// CosformerModel_39273180954971
// MI455X (gfx1250) — matrix operand order repaired and checked by reading, not compiled, not run on hardware
//
#include <hip/hip_runtime.h>

// ---------------------------------------------------------------------------
// CDNA5 (gfx1250) cosformer forward: LDS-free bf16 WMMA pipeline.
// K loops are unrolled x2 with ping-pong fragment buffers: no rotation
// copies, loads for step k+1 overlap the WMMAs of step k.
// ---------------------------------------------------------------------------

typedef __bf16 bf16;
typedef bf16 v16bf __attribute__((ext_vector_type(16)));
typedef bf16 v8bf  __attribute__((ext_vector_type(8)));
typedef float v8f  __attribute__((ext_vector_type(8)));

constexpr int CB = 8, CL = 4096, CD = 512, CH = 8, CDH = 64, CDEPTH = 4;
constexpr int CD4 = 2048, CNC = 2;
constexpr int CM = CB * CL;             // 32768 rows
constexpr int CKVP = 16;                // L partitions for k^T v
constexpr float CEPS = 1e-5f;

// A-fragment (16x32 bf16, MxK): lane holds row m=lane&15; elements j=0..7 at
// k=kb+j, j=8..15 at k=kb+16+(j-8), kb = (lane>=16 ? 8 : 0).
static __device__ __forceinline__ v16bf loadA16(const bf16* p, int kb) {
  v8bf lo = *(const v8bf*)(p + kb);
  v8bf hi = *(const v8bf*)(p + kb + 16);
  return __builtin_shufflevector(lo, hi, 0, 1, 2, 3, 4, 5, 6, 7,
                                 8, 9, 10, 11, 12, 13, 14, 15);
}

// B-fragment (32x16 bf16, KxN): lane holds col n=lane&15; elements j=0..7 at
// k=kb+j, j=8..15 at k=kb+16+(j-8), kb = (lane>=16 ? 8 : 0).  Operand stored (N,K) row-major.
static __device__ __forceinline__ v16bf loadB16(const bf16* p, int kb) {
  v8bf lo = *(const v8bf*)(p + kb);
  v8bf hi = *(const v8bf*)(p + kb + 16);
  return __builtin_shufflevector(lo, hi, 0, 1, 2, 3, 4, 5, 6, 7,
                                 8, 9, 10, 11, 12, 13, 14, 15);
}

static __device__ __forceinline__ v8f wmma_bf16(v16bf a, v16bf b, v8f c) {
  return __builtin_amdgcn_wmma_f32_16x16x32_bf16(false, a, false, b,
                                                 (short)0, c, false, false);
}

// ---------------------------------------------------------------------------
// Elementwise helpers
// ---------------------------------------------------------------------------

__global__ __launch_bounds__(256)
void cvt_f32_bf16_kernel(const float* __restrict__ in, bf16* __restrict__ out,
                         size_t n) {
  size_t i = (size_t)blockIdx.x * blockDim.x + threadIdx.x;
  if (i < n) out[i] = (bf16)in[i];
}

// Transpose-convert one weight matrix: wt[n*K+k] = (bf16)w[k*N+n]
__global__ __launch_bounds__(256)
void cvtT_kernel(const float* __restrict__ w, bf16* __restrict__ wt,
                 int K, int N) {
  size_t i = (size_t)blockIdx.x * blockDim.x + threadIdx.x;
  if (i >= (size_t)K * N) return;
  int n = (int)(i % N);
  int k = (int)(i / N);
  wt[(size_t)n * K + k] = (bf16)w[i];
}

__global__ __launch_bounds__(256)
void embed_kernel(const int* __restrict__ x, const float* __restrict__ emb,
                  float* __restrict__ h) {
  size_t i = (size_t)blockIdx.x * blockDim.x + threadIdx.x;
  if (i >= (size_t)CM * CD) return;
  size_t bl = i / CD;
  int d = (int)(i % CD);
  h[i] = emb[(size_t)x[bl] * CD + d];
}

__global__ __launch_bounds__(256)
void trig_kernel(const int* __restrict__ lengths, float* __restrict__ cosb,
                 float* __restrict__ sinb, float* __restrict__ maskb) {
  int i = blockIdx.x * blockDim.x + threadIdx.x;
  if (i >= CB * CL) return;
  int b = i / CL, l = i % CL;
  float len = (float)lengths[b];
  float ang = 1.5707963267948966f * (float)(l + 1) / len;
  cosb[i] = __cosf(ang);
  sinb[i] = __sinf(ang);
  maskb[i] = (l < lengths[b]) ? 1.0f : 0.0f;
}

// q -> (q*cos, q*sin), (B,L,D) layout. outs may alias in.
__global__ __launch_bounds__(256)
void rope_kernel(const bf16* in, bf16* outc, bf16* outs,
                 const float* __restrict__ cosb, const float* __restrict__ sinb) {
  size_t i = (size_t)blockIdx.x * blockDim.x + threadIdx.x;
  if (i >= (size_t)CM * CD) return;
  size_t bl = i / CD;
  float v = (float)in[i];
  float c = cosb[bl], s = sinb[bl];
  bf16 vc = (bf16)(v * c);
  bf16 vs = (bf16)(v * s);
  outc[i] = vc;
  outs[i] = vs;
}

// k (B,L,D) -> kc^T, ks^T in (B,H,DH,L) layout (K-contiguous per (d) row).
__global__ __launch_bounds__(256)
void ropeT_kernel(const bf16* __restrict__ in, bf16* __restrict__ outcT,
                  bf16* __restrict__ outsT, const float* __restrict__ cosb,
                  const float* __restrict__ sinb) {
  size_t i = (size_t)blockIdx.x * blockDim.x + threadIdx.x;  // (b,h,dh,l)
  if (i >= (size_t)CM * CD) return;
  int l = (int)(i % CL);
  size_t r = i / CL;
  int dh = (int)(r % CDH);
  size_t bh = r / CDH;
  int hh = (int)(bh % CH);
  int b = (int)(bh / CH);
  float v = (float)in[((size_t)b * CL + l) * CD + hh * CDH + dh];
  float c = cosb[b * CL + l], s = sinb[b * CL + l];
  outcT[i] = (bf16)(v * c);
  outsT[i] = (bf16)(v * s);
}

// v (B,L,D) -> v^T (B,H,DH,L)
__global__ __launch_bounds__(256)
void transT_kernel(const bf16* __restrict__ in, bf16* __restrict__ outT) {
  size_t i = (size_t)blockIdx.x * blockDim.x + threadIdx.x;
  if (i >= (size_t)CM * CD) return;
  int l = (int)(i % CL);
  size_t r = i / CL;
  int dh = (int)(r % CDH);
  size_t bh = r / CDH;
  int hh = (int)(bh % CH);
  int b = (int)(bh / CH);
  outT[i] = in[((size_t)b * CL + l) * CD + hh * CDH + dh];
}

// ksum[bh, sel, dh] = sum_l kcT/ksT row (rows are contiguous in l)
__global__ __launch_bounds__(256)
void ksumT_kernel(const bf16* __restrict__ kcT, const bf16* __restrict__ ksT,
                  float* __restrict__ ksum) {
  int i = blockIdx.x * blockDim.x + threadIdx.x;  // B*H*DH
  if (i >= CB * CH * CDH) return;
  const bf16* pc = kcT + (size_t)i * CL;
  const bf16* ps = ksT + (size_t)i * CL;
  float sc = 0.f, ss = 0.f;
  for (int l = 0; l < CL; l += 8) {
    v8bf xc = *(const v8bf*)(pc + l);
    v8bf xs = *(const v8bf*)(ps + l);
#pragma unroll
    for (int j = 0; j < 8; ++j) {
      sc += (float)xc[j];
      ss += (float)xs[j];
    }
  }
  int dh = i & 63, bh = i >> 6;
  ksum[((size_t)bh * 2 + 0) * CDH + dh] = sc;
  ksum[((size_t)bh * 2 + 1) * CDH + dh] = ss;
}

// den[b,l,h] = qc . ksum_c + qs . ksum_s
__global__ __launch_bounds__(256)
void den_kernel(const bf16* __restrict__ qc, const bf16* __restrict__ qs,
                const float* __restrict__ ksum, float* __restrict__ den) {
  int i = blockIdx.x * blockDim.x + threadIdx.x;
  if (i >= CB * CL * CH) return;
  int hh = i % CH;
  int bl = i / CH;
  int b = bl / CL;
  const bf16* qcp = qc + (size_t)bl * CD + hh * CDH;
  const bf16* qsp = qs + (size_t)bl * CD + hh * CDH;
  const float* kcs = ksum + (((size_t)(b * CH + hh)) * 2 + 0) * CDH;
  const float* kss = ksum + (((size_t)(b * CH + hh)) * 2 + 1) * CDH;
  float s = 0.f;
  for (int d0 = 0; d0 < CDH; d0 += 8) {
    v8bf xc = *(const v8bf*)(qcp + d0);
    v8bf xs = *(const v8bf*)(qsp + d0);
#pragma unroll
    for (int j = 0; j < 8; ++j)
      s += (float)xc[j] * kcs[d0 + j] + (float)xs[j] * kss[d0 + j];
  }
  den[i] = s;
}

__global__ __launch_bounds__(256)
void pool_kernel(const float* __restrict__ h, float* __restrict__ pooled) {
  int i = blockIdx.x * blockDim.x + threadIdx.x;  // B*D
  if (i >= CB * CD) return;
  int b = i / CD, d = i % CD;
  float s = 0.f;
  for (int l = 0; l < CL; ++l) s += h[((size_t)b * CL + l) * CD + d];
  pooled[i] = s * (1.0f / (float)CL);
}

__global__ __launch_bounds__(32)
void final_kernel(const float* __restrict__ pooled, const float* __restrict__ Wf,
                  const float* __restrict__ bfv, float* __restrict__ out) {
  int i = threadIdx.x;  // B*NCLASS = 16
  if (i >= CB * CNC) return;
  int b = i / CNC, c = i % CNC;
  float s = bfv[c];
  for (int d = 0; d < CD; ++d) s += pooled[b * CD + d] * Wf[d * CNC + c];
  out[i] = s;
}

// ---------------------------------------------------------------------------
// Main GEMM: C(M,N) = A(M,K) @ Wt(N,K)^T, all operands K-contiguous.
// Block = 8 waves, tile 64(M) x 256(N); wave 32x64: 8 accumulators.
// Ping-pong pipelined K loop (unroll x2, no buffer-rotation copies).
// Requires K % 64 == 0 (K = 512 or 2048 here).
// ---------------------------------------------------------------------------
__global__ __launch_bounds__(256)
void gemm_bf16_kernel(const bf16* __restrict__ A, const bf16* __restrict__ Wt,
                      int M, int N, int K,
                      bf16* __restrict__ outb, float* __restrict__ outf,
                      const float* __restrict__ res,
                      const float* __restrict__ scale_ptr,
                      const float* __restrict__ rowscale, int relu) {
  const int tid = threadIdx.x, lane = tid & 31, wave = tid >> 5;
  const int nBN = N >> 8;
  const int bm = (int)(blockIdx.x / nBN) * 64;
  const int bn = (int)(blockIdx.x % nBN) * 256;
  const int wm = (wave >> 2) << 5;   // 0 / 32
  const int wn = (wave & 3) << 6;    // 0..192
  const int lr = lane & 15;
  const int kbA = (lane >> 4) ? 8 : 0;
  const int kbB = (lane >> 4) ? 8 : 0;

  const bf16* arow0 = A + (size_t)(bm + wm + lr) * K;
  const bf16* arow1 = arow0 + (size_t)16 * K;
  const bf16* brow = Wt + (size_t)(bn + wn + lr) * K;

  v8f acc[2][4] = {};

  v16bf a0A, a1A, bA[4];   // ping buffer
  v16bf a0B, a1B, bB[4];   // pong buffer

  // Prologue: buffer A holds k = 0.
  a0A = loadA16(arow0, kbA);
  a1A = loadA16(arow1, kbA);
#pragma unroll
  for (int t = 0; t < 4; ++t) bA[t] = loadB16(brow + (size_t)(t * 16) * K, kbB);

  int k0 = 32;
  for (; k0 < K - 32; k0 += 64) {
    // Load pong @ k0 while computing ping (k0-32).
    a0B = loadA16(arow0 + k0, kbA);
    a1B = loadA16(arow1 + k0, kbA);
#pragma unroll
    for (int t = 0; t < 4; ++t)
      bB[t] = loadB16(brow + (size_t)(t * 16) * K + k0, kbB);
#pragma unroll
    for (int t = 0; t < 4; ++t) {
      acc[0][t] = wmma_bf16(a0A, bA[t], acc[0][t]);
      acc[1][t] = wmma_bf16(a1A, bA[t], acc[1][t]);
    }
    // Load ping @ k0+32 while computing pong (k0).
    a0A = loadA16(arow0 + k0 + 32, kbA);
    a1A = loadA16(arow1 + k0 + 32, kbA);
#pragma unroll
    for (int t = 0; t < 4; ++t)
      bA[t] = loadB16(brow + (size_t)(t * 16) * K + k0 + 32, kbB);
#pragma unroll
    for (int t = 0; t < 4; ++t) {
      acc[0][t] = wmma_bf16(a0B, bB[t], acc[0][t]);
      acc[1][t] = wmma_bf16(a1B, bB[t], acc[1][t]);
    }
  }
  // Tail: k0 == K-32; ping holds K-64.
  a0B = loadA16(arow0 + k0, kbA);
  a1B = loadA16(arow1 + k0, kbA);
#pragma unroll
  for (int t = 0; t < 4; ++t)
    bB[t] = loadB16(brow + (size_t)(t * 16) * K + k0, kbB);
#pragma unroll
  for (int t = 0; t < 4; ++t) {
    acc[0][t] = wmma_bf16(a0A, bA[t], acc[0][t]);
    acc[1][t] = wmma_bf16(a1A, bA[t], acc[1][t]);
  }
#pragma unroll
  for (int t = 0; t < 4; ++t) {
    acc[0][t] = wmma_bf16(a0B, bB[t], acc[0][t]);
    acc[1][t] = wmma_bf16(a1B, bB[t], acc[1][t]);
  }

  const float scale = scale_ptr ? *scale_ptr : 1.0f;
#pragma unroll
  for (int s = 0; s < 2; ++s) {
#pragma unroll
    for (int t = 0; t < 4; ++t) {
      const int n = bn + wn + t * 16 + lr;
#pragma unroll
      for (int r = 0; r < 8; ++r) {
        int m = bm + wm + s * 16 + r + ((lane >> 4) << 3);
        float v = acc[s][t][r];
        if (relu) v = fmaxf(v, 0.f);
        if (rowscale) v *= rowscale[m];
        size_t o = (size_t)m * N + n;
        if (outb) outb[o] = (bf16)v;
        else      outf[o] = res[o] + scale * v;
      }
    }
  }
}

// ---------------------------------------------------------------------------
// Partial k^T v: kvpart[part][bh][sel][d][e] over an L/CKVP slice (8 steps
// of 32, even).  Ping-pong pipelined like the main GEMM.
// ---------------------------------------------------------------------------
__global__ __launch_bounds__(256)
void kvpart_kernel(const bf16* __restrict__ kcT, const bf16* __restrict__ ksT,
                   const bf16* __restrict__ vT, float* __restrict__ kvpart) {
  const int blk = blockIdx.x;            // bh * CKVP + part
  const int part = blk % CKVP;
  const int bh = blk / CKVP;
  const int tid = threadIdx.x, lane = tid & 31, wave = tid >> 5;
  const int sel = wave >> 2, dm = wave & 3;
  const int lr = lane & 15;
  const int kbA = (lane >> 4) ? 8 : 0;
  const int kbB = (lane >> 4) ? 8 : 0;

  const bf16* aT = (sel ? ksT : kcT) + ((size_t)bh * CDH + dm * 16 + lr) * CL;
  const bf16* bT = vT + ((size_t)bh * CDH + lr) * CL;

  const int l0beg = part * (CL / CKVP);
  const int l0end = l0beg + CL / CKVP;

  v8f acc[4] = {};
  v16bf aA, bA[4], aB, bB[4];

  aA = loadA16(aT + l0beg, kbA);
#pragma unroll
  for (int t = 0; t < 4; ++t)
    bA[t] = loadB16(bT + (size_t)(t * 16) * CL + l0beg, kbB);

  int l0 = l0beg + 32;
  for (; l0 < l0end - 32; l0 += 64) {
    aB = loadA16(aT + l0, kbA);
#pragma unroll
    for (int t = 0; t < 4; ++t)
      bB[t] = loadB16(bT + (size_t)(t * 16) * CL + l0, kbB);
#pragma unroll
    for (int t = 0; t < 4; ++t) acc[t] = wmma_bf16(aA, bA[t], acc[t]);

    aA = loadA16(aT + l0 + 32, kbA);
#pragma unroll
    for (int t = 0; t < 4; ++t)
      bA[t] = loadB16(bT + (size_t)(t * 16) * CL + l0 + 32, kbB);
#pragma unroll
    for (int t = 0; t < 4; ++t) acc[t] = wmma_bf16(aB, bB[t], acc[t]);
  }
  aB = loadA16(aT + l0, kbA);
#pragma unroll
  for (int t = 0; t < 4; ++t)
    bB[t] = loadB16(bT + (size_t)(t * 16) * CL + l0, kbB);
#pragma unroll
  for (int t = 0; t < 4; ++t) acc[t] = wmma_bf16(aA, bA[t], acc[t]);
#pragma unroll
  for (int t = 0; t < 4; ++t) acc[t] = wmma_bf16(aB, bB[t], acc[t]);

  float* outp = kvpart + (((size_t)part * CB * CH + bh) * 2 + sel) * CDH * CDH;
#pragma unroll
  for (int t = 0; t < 4; ++t) {
    int e = t * 16 + lr;
#pragma unroll
    for (int r = 0; r < 8; ++r) {
      int d = dm * 16 + r + ((lane >> 4) << 3);
      outp[d * CDH + e] = acc[t][r];
    }
  }
}

// Reduce partials and store transposed bf16: kvT[bh][sel][e][d]
__global__ __launch_bounds__(256)
void kvreduce_kernel(const float* __restrict__ kvpart, bf16* __restrict__ kvT) {
  int i = blockIdx.x * blockDim.x + threadIdx.x;  // (bh*2+sel, d, e)
  if (i >= CB * CH * 2 * CDH * CDH) return;
  int e = i & 63, d = (i >> 6) & 63, rest = i >> 12;
  size_t stride = (size_t)CB * CH * 2 * CDH * CDH;
  float s = 0.f;
#pragma unroll
  for (int p = 0; p < CKVP; ++p) s += kvpart[(size_t)p * stride + i];
  kvT[((size_t)rest * CDH + e) * CDH + d] = (bf16)s;
}

// ---------------------------------------------------------------------------
// attn = (qc@kvc + qs@kvs) / (den + eps), bf16 out in (B,L,D).
// Two stages (cos, sin); each stage loads all 10 fragments, then 8 WMMAs.
// grid = B*H*(L/128); wave: 16 l x 64 e.
// ---------------------------------------------------------------------------
__global__ __launch_bounds__(256)
void attn_kernel(const bf16* __restrict__ qc, const bf16* __restrict__ qs,
                 const bf16* __restrict__ kvT, const float* __restrict__ den,
                 bf16* __restrict__ outattn) {
  const int nLT = CL / 128;
  const int bh = (int)(blockIdx.x / nLT), lt = (int)(blockIdx.x % nLT);
  const int b = bh / CH, hh = bh % CH;
  const int tid = threadIdx.x, lane = tid & 31, wave = tid >> 5;
  const int lr = lane & 15;
  const int l0 = lt * 128 + wave * 16;
  const int kbA = (lane >> 4) ? 8 : 0;
  const int kbB = (lane >> 4) ? 8 : 0;

  const bf16* qcrow = qc + ((size_t)b * CL + l0 + lr) * CD + hh * CDH;
  const bf16* qsrow = qs + ((size_t)b * CL + l0 + lr) * CD + hh * CDH;
  const bf16* kvc = kvT + ((size_t)bh * 2 + 0) * CDH * CDH;
  const bf16* kvs = kvT + ((size_t)bh * 2 + 1) * CDH * CDH;

  v8f acc[4] = {};

  // Stage 1: qc x kvc (load all fragments, then compute).
  {
    v16bf a0 = loadA16(qcrow, kbA);
    v16bf a1 = loadA16(qcrow + 32, kbA);
    v16bf b0[4], b1[4];
#pragma unroll
    for (int t = 0; t < 4; ++t) {
      b0[t] = loadB16(kvc + (size_t)(t * 16 + lr) * CDH, kbB);
      b1[t] = loadB16(kvc + (size_t)(t * 16 + lr) * CDH + 32, kbB);
    }
#pragma unroll
    for (int t = 0; t < 4; ++t) acc[t] = wmma_bf16(a0, b0[t], acc[t]);
#pragma unroll
    for (int t = 0; t < 4; ++t) acc[t] = wmma_bf16(a1, b1[t], acc[t]);
  }
  // Stage 2: qs x kvs.
  {
    v16bf a0 = loadA16(qsrow, kbA);
    v16bf a1 = loadA16(qsrow + 32, kbA);
    v16bf b0[4], b1[4];
#pragma unroll
    for (int t = 0; t < 4; ++t) {
      b0[t] = loadB16(kvs + (size_t)(t * 16 + lr) * CDH, kbB);
      b1[t] = loadB16(kvs + (size_t)(t * 16 + lr) * CDH + 32, kbB);
    }
#pragma unroll
    for (int t = 0; t < 4; ++t) acc[t] = wmma_bf16(a0, b0[t], acc[t]);
#pragma unroll
    for (int t = 0; t < 4; ++t) acc[t] = wmma_bf16(a1, b1[t], acc[t]);
  }

  float dn[8];
#pragma unroll
  for (int r = 0; r < 8; ++r) {
    int m = l0 + r + ((lane >> 4) << 3);
    dn[r] = den[((size_t)b * CL + m) * CH + hh] + CEPS;
  }
#pragma unroll
  for (int t = 0; t < 4; ++t) {
    int n = t * 16 + lr;
#pragma unroll
    for (int r = 0; r < 8; ++r) {
      int m = l0 + r + ((lane >> 4) << 3);
      outattn[((size_t)b * CL + m) * CD + hh * CDH + n] = (bf16)(acc[t][r] / dn[r]);
    }
  }
}

// ---------------------------------------------------------------------------
// Host launcher
// ---------------------------------------------------------------------------
extern "C" void kernel_launch(void* const* d_in, const int* in_sizes, int n_in,
                              void* d_out, int out_size, void* d_ws, size_t ws_size,
                              hipStream_t stream) {
  (void)in_sizes; (void)n_in; (void)out_size; (void)ws_size;

  const int*   x       = (const int*)d_in[0];
  const int*   lengths = (const int*)d_in[1];
  const float* emb     = (const float*)d_in[2];
  const float* Wq      = (const float*)d_in[3];
  const float* Wk      = (const float*)d_in[4];
  const float* Wv      = (const float*)d_in[5];
  const float* Wo      = (const float*)d_in[6];
  const float* W1      = (const float*)d_in[7];
  const float* W2      = (const float*)d_in[8];
  const float* rw      = (const float*)d_in[9];
  const float* Wf      = (const float*)d_in[10];
  const float* bfv     = (const float*)d_in[11];
  float* out = (float*)d_out;

  char* ws = (char*)d_ws;
  size_t off = 0;
  auto alloc = [&](size_t bytes) -> char* {
    char* p = ws + off;
    off += (bytes + 255) & ~(size_t)255;
    return p;
  };

  const size_t szDD  = (size_t)CD * CD;
  const size_t szD1  = (size_t)CD * CD4;
  const size_t nBLD  = (size_t)CM * CD;
  const size_t nBLD4 = (size_t)CM * CD4;

  bf16* WqT = (bf16*)alloc(CDEPTH * szDD * 2);  // (N,K) transposed bf16
  bf16* WkT = (bf16*)alloc(CDEPTH * szDD * 2);
  bf16* WvT = (bf16*)alloc(CDEPTH * szDD * 2);
  bf16* WoT = (bf16*)alloc(CDEPTH * szDD * 2);
  bf16* W1T = (bf16*)alloc(CDEPTH * szD1 * 2);
  bf16* W2T = (bf16*)alloc(CDEPTH * szD1 * 2);
  float* h  = (float*)alloc(nBLD * 4);
  bf16* R1  = (bf16*)alloc(nBLD * 2);   // hb
  bf16* R2  = (bf16*)alloc(nBLD * 2);   // qb -> qs
  bf16* R3  = (bf16*)alloc(nBLD * 2);   // kb -> attn
  bf16* R4  = (bf16*)alloc(nBLD * 2);   // vb (masked)
  bf16* R5  = (bf16*)alloc(nBLD * 2);   // qc
  bf16* R0  = (bf16*)alloc(nBLD4 * 2);  // kcT|ksT|vT, later ffn1
  float* kvpartb = (float*)alloc((size_t)CKVP * CB * CH * 2 * CDH * CDH * 4);
  bf16* kvTb  = (bf16*)alloc((size_t)CB * CH * 2 * CDH * CDH * 2);
  float* ksumb = (float*)alloc((size_t)CB * CH * 2 * CDH * 4);
  float* cosb  = (float*)alloc((size_t)CB * CL * 4);
  float* sinb  = (float*)alloc((size_t)CB * CL * 4);
  float* maskb = (float*)alloc((size_t)CB * CL * 4);
  float* denb  = (float*)alloc((size_t)CB * CL * CH * 4);
  float* pooled = (float*)alloc((size_t)CB * CD * 4);

  bf16* kcT = R0;
  bf16* ksT = R0 + nBLD;
  bf16* vT  = R0 + 2 * nBLD;

  auto g1 = [](size_t n) { return dim3((unsigned)((n + 255) / 256)); };

  for (int i = 0; i < CDEPTH; ++i) {
    cvtT_kernel<<<g1(szDD), 256, 0, stream>>>(Wq + i * szDD, WqT + i * szDD, CD, CD);
    cvtT_kernel<<<g1(szDD), 256, 0, stream>>>(Wk + i * szDD, WkT + i * szDD, CD, CD);
    cvtT_kernel<<<g1(szDD), 256, 0, stream>>>(Wv + i * szDD, WvT + i * szDD, CD, CD);
    cvtT_kernel<<<g1(szDD), 256, 0, stream>>>(Wo + i * szDD, WoT + i * szDD, CD, CD);
    cvtT_kernel<<<g1(szD1), 256, 0, stream>>>(W1 + i * szD1, W1T + i * szD1, CD, CD4);
    cvtT_kernel<<<g1(szD1), 256, 0, stream>>>(W2 + i * szD1, W2T + i * szD1, CD4, CD);
  }

  trig_kernel<<<g1((size_t)CB * CL), 256, 0, stream>>>(lengths, cosb, sinb, maskb);
  embed_kernel<<<g1(nBLD), 256, 0, stream>>>(x, emb, h);

  const dim3 gridDD((CM / 64) * (CD / 256));    // N=512
  const dim3 gridD4((CM / 64) * (CD4 / 256));   // N=2048

  for (int i = 0; i < CDEPTH; ++i) {
    const bf16* wq = WqT + i * szDD;
    const bf16* wk = WkT + i * szDD;
    const bf16* wv = WvT + i * szDD;
    const bf16* wo = WoT + i * szDD;
    const bf16* w1 = W1T + i * szD1;
    const bf16* w2 = W2T + i * szD1;

    cvt_f32_bf16_kernel<<<g1(nBLD), 256, 0, stream>>>(h, R1, nBLD);

    // q = relu(h@Wq); k = relu(h@Wk)*mask; v = (h@Wv)*mask
    gemm_bf16_kernel<<<gridDD, 256, 0, stream>>>(R1, wq, CM, CD, CD, R2, nullptr,
                                                 nullptr, nullptr, nullptr, 1);
    gemm_bf16_kernel<<<gridDD, 256, 0, stream>>>(R1, wk, CM, CD, CD, R3, nullptr,
                                                 nullptr, nullptr, maskb, 1);
    gemm_bf16_kernel<<<gridDD, 256, 0, stream>>>(R1, wv, CM, CD, CD, R4, nullptr,
                                                 nullptr, nullptr, maskb, 0);

    // qc -> R5, qs -> R2 (in place); kcT/ksT, vT in transposed layout
    rope_kernel<<<g1(nBLD), 256, 0, stream>>>(R2, R5, R2, cosb, sinb);
    ropeT_kernel<<<g1(nBLD), 256, 0, stream>>>(R3, kcT, ksT, cosb, sinb);
    transT_kernel<<<g1(nBLD), 256, 0, stream>>>(R4, vT);

    ksumT_kernel<<<g1((size_t)CB * CH * CDH), 256, 0, stream>>>(kcT, ksT, ksumb);
    kvpart_kernel<<<dim3(CB * CH * CKVP), 256, 0, stream>>>(kcT, ksT, vT, kvpartb);
    kvreduce_kernel<<<g1((size_t)CB * CH * 2 * CDH * CDH), 256, 0, stream>>>(kvpartb, kvTb);
    den_kernel<<<g1((size_t)CB * CL * CH), 256, 0, stream>>>(R5, R2, ksumb, denb);

    // attn (bf16) -> R3 (kb dead)
    attn_kernel<<<dim3(CB * CH * (CL / 128)), 256, 0, stream>>>(R5, R2, kvTb,
                                                                denb, R3);

    // h += rw[i,0] * (attn @ Wo)
    gemm_bf16_kernel<<<gridDD, 256, 0, stream>>>(R3, wo, CM, CD, CD, nullptr, h,
                                                 h, rw + i * 2 + 0, nullptr, 0);

    // FFN
    cvt_f32_bf16_kernel<<<g1(nBLD), 256, 0, stream>>>(h, R1, nBLD);
    gemm_bf16_kernel<<<gridD4, 256, 0, stream>>>(R1, w1, CM, CD4, CD, R0, nullptr,
                                                 nullptr, nullptr, nullptr, 1);
    gemm_bf16_kernel<<<gridDD, 256, 0, stream>>>(R0, w2, CM, CD, CD4, nullptr, h,
                                                 h, rw + i * 2 + 1, nullptr, 0);
  }

  pool_kernel<<<g1((size_t)CB * CD), 256, 0, stream>>>(h, pooled);
  final_kernel<<<dim3(1), 32, 0, stream>>>(pooled, Wf, bfv, out);
}
